// DigitConvolutionalModel_67774583931474
// MI455X (gfx1250) — compile-verified, matrix operand order checked by reading, not run on hardware
//
#include <hip/hip_runtime.h>

typedef __attribute__((ext_vector_type(16))) _Float16 v16h;
typedef __attribute__((ext_vector_type(8)))  float    v8f;

#define BATCH 16384
#define IMG   28
#define OUTW  26
#define KPIX  676            // 26*26 conv outputs
#define KPAD  704            // padded to 22*32
#define KT1   22             // K-tiles for GEMM1
#define HID   512
#define KT2   16             // K-tiles for GEMM2 (512/32)
#define NCLS  10

// ---------------------------------------------------------------------------
// Fragment-layout helpers (CDNA5 ISA, 16-bit A/B 16x32):
//   lane < 16 : holds K%32 in {0..7} (half j=0..7) and {16..23} (half j=8..15)
//   lane >= 16: holds K%32 in {8..15}          and {24..31}
// A-frag flat index for element (row, k):
//   (((rowTile*KT + kTile)*32 + lane)*16 + j),  lane = row%16 | hiBit<<4
// B-frag identical with column index playing the role of row.
// ---------------------------------------------------------------------------

// conv(3x3 valid) on x[B,784], output written directly as f16 A-fragments,
// K padded 676 -> 704 with zeros.  blockDim = (64,4): x = conv index chunk,
// y = batch row within block.
__global__ void conv_to_afrag(const float* __restrict__ x,
                              const float* __restrict__ cw,
                              _Float16* __restrict__ afrag) {
    const int k = blockIdx.x * 64 + threadIdx.x;          // 0..703
    const int m = blockIdx.y * 4 + threadIdx.y;           // batch row
    float v = 0.0f;
    if (k < KPIX) {
        const int r = k / OUTW;
        const int c = k - r * OUTW;
        const float* xp = x + (size_t)m * (IMG * IMG) + r * IMG + c;
#pragma unroll
        for (int i = 0; i < 3; ++i)
#pragma unroll
            for (int j = 0; j < 3; ++j)
                v += xp[i * IMG + j] * cw[i * 3 + j];
    }
    const int mt   = m >> 4;
    const int kt   = k >> 5;
    const int kp   = k & 31;
    const int lane = (m & 15) | (((kp >> 3) & 1) << 4);
    const int j    = (kp & 7) | (((kp >> 4) & 1) << 3);
    afrag[((((size_t)mt * KT1 + kt) * 32 + lane) << 4) + j] = (_Float16)v;
}

// W1[676,512] f32 -> B-fragments [nt][kt][lane][16] f16, K padded to 704.
__global__ void pack_w1(const float* __restrict__ W1, _Float16* __restrict__ w1f) {
    const int tid = blockIdx.x * 256 + threadIdx.x;  // 704*512 threads
    const int n = tid & (HID - 1);
    const int k = tid >> 9;                          // 0..703
    const float v = (k < KPIX) ? W1[(size_t)k * HID + n] : 0.0f;
    const int nt   = n >> 4;
    const int kt   = k >> 5;
    const int kp   = k & 31;
    const int lane = (n & 15) | (((kp >> 3) & 1) << 4);
    const int j    = (kp & 7) | (((kp >> 4) & 1) << 3);
    w1f[((((size_t)nt * KT1 + kt) * 32 + lane) << 4) + j] = (_Float16)v;
}

// W2[512,10] f32 -> B-fragments [kt][lane][16] f16, N padded to 16.
__global__ void pack_w2(const float* __restrict__ W2, _Float16* __restrict__ w2f) {
    const int tid = blockIdx.x * 256 + threadIdx.x;  // 512*16 threads
    const int n = tid & 15;
    const int k = tid >> 4;                          // 0..511
    const float v = (n < NCLS) ? W2[(size_t)k * NCLS + n] : 0.0f;
    const int kt   = k >> 5;
    const int kp   = k & 31;
    const int lane = (n & 15) | (((kp >> 3) & 1) << 4);
    const int j    = (kp & 7) | (((kp >> 4) & 1) << 3);
    w2f[(((kt * 32) + lane) << 4) + j] = (_Float16)v;
}

// Fused: h = relu(a @ W1 + b1) kept in LDS (f16, A-frag layout), then
// logits = h @ W2 + b2 written to d_out.
// 256 threads = 8 waves; block owns 128 batch rows (8 m-tiles).
// GEMM1 wave tiling: waves arranged 4 (m) x 2 (n); each wave computes a
// 2 m-tile x 4 n-tile patch -> 8 WMMA accumulators, 6 fragment loads per
// k-step (2 A + 4 B) instead of 9.
__global__ void __launch_bounds__(256)
mlp_fused(const _Float16* __restrict__ afrag,
          const _Float16* __restrict__ w1f,
          const float* __restrict__ b1,
          const _Float16* __restrict__ w2f,
          const float* __restrict__ b2,
          float* __restrict__ out) {
    __shared__ _Float16 hbuf[8 * KT2 * 32 * 16];   // 128 KB: h[128,512] as A-frags

    const int lane = threadIdx.x & 31;
    const int wave = threadIdx.x >> 5;             // 0..7
    const int wm   = wave >> 1;                    // 0..3 (m-group)
    const int wn   = wave & 1;                     // 0..1 (n-group)
    const int hi   = lane >> 4;                    // 0/1 half-wave
    const int nlo  = lane & 15;

    // ---------------- GEMM1 + bias + ReLU, 4 chunks of 128 hidden cols ------
    for (int chunk = 0; chunk < 4; ++chunk) {
        v8f acc[2][4] = {};
        for (int kt = 0; kt < KT1; ++kt) {
            v16h a[2];
#pragma unroll
            for (int mi = 0; mi < 2; ++mi) {
                const size_t mt = (size_t)blockIdx.x * 8 + wm * 2 + mi;
                a[mi] = *(const v16h*)(afrag + (((mt * KT1 + kt) * 32 + lane) << 4));
            }
#pragma unroll
            for (int ni = 0; ni < 4; ++ni) {
                const size_t nt = chunk * 8 + wn * 4 + ni;
                const v16h b = *(const v16h*)(w1f +
                    (((nt * KT1 + kt) * 32 + lane) << 4));
#pragma unroll
                for (int mi = 0; mi < 2; ++mi)
                    acc[mi][ni] = __builtin_amdgcn_wmma_f32_16x16x32_f16(
                        false, a[mi], false, b, (short)0, acc[mi][ni], false, false);
            }
        }
        // C/D layout -> bias/ReLU -> f16 A-frag layout in LDS (k2 = hidden idx)
#pragma unroll
        for (int ni = 0; ni < 4; ++ni) {
            const int n    = chunk * 128 + (wn * 4 + ni) * 16 + nlo; // hidden idx
            const float bs = b1[n];
            const int kt2   = n >> 5;
            const int kp    = n & 31;
            const int l2bit = ((kp >> 3) & 1) << 4;
            const int j2    = (kp & 7) | (((kp >> 4) & 1) << 3);
#pragma unroll
            for (int mi = 0; mi < 2; ++mi) {
                const int mt2 = wm * 2 + mi;                 // local m-tile
#pragma unroll
                for (int j = 0; j < 8; ++j) {
                    const int mrow = j + (hi << 3);          // row within tile
                    float hv = acc[mi][ni][j] + bs;
                    hv = hv > 0.0f ? hv : 0.0f;
                    const int lane2 = mrow | l2bit;
                    hbuf[(((mt2 * KT2 + kt2) * 32 + lane2) << 4) + j2] =
                        (_Float16)hv;
                }
            }
        }
    }
    __syncthreads();

    // ---------------- GEMM2: one 16x16 tile per wave over K=512 -------------
    v8f acc2 = {};
    for (int kt = 0; kt < KT2; ++kt) {
        const v16h a = *(const v16h*)(hbuf + (((wave * KT2 + kt) * 32 + lane) << 4));
        const v16h b = *(const v16h*)(w2f + (((kt * 32) + lane) << 4));
        acc2 = __builtin_amdgcn_wmma_f32_16x16x32_f16(
            false, a, false, b, (short)0, acc2, false, false);
    }
    if (nlo < NCLS) {
        const float bs = b2[nlo];
#pragma unroll
        for (int j = 0; j < 8; ++j) {
            const int m = blockIdx.x * 128 + wave * 16 + j + (hi << 3);
            out[(size_t)m * NCLS + nlo] = acc2[j] + bs;
        }
    }
}

extern "C" void kernel_launch(void* const* d_in, const int* in_sizes, int n_in,
                              void* d_out, int out_size, void* d_ws, size_t ws_size,
                              hipStream_t stream) {
    const float* x  = (const float*)d_in[0];   // [16384, 784]
    const float* cw = (const float*)d_in[1];   // [3, 3]
    const float* W1 = (const float*)d_in[2];   // [676, 512]
    const float* b1 = (const float*)d_in[3];   // [512]
    const float* W2 = (const float*)d_in[4];   // [512, 10]
    const float* b2 = (const float*)d_in[5];   // [10]
    float* out = (float*)d_out;                // [16384, 10]

    // workspace layout (f16): afrag | w1f | w2f  (all 32B aligned)
    _Float16* afrag = (_Float16*)d_ws;                       // 16384*704
    _Float16* w1f   = afrag + (size_t)BATCH * KPAD;          // 704*512
    _Float16* w2f   = w1f + (size_t)KPAD * HID;              // 512*16

    pack_w1<<<(KPAD * HID) / 256, 256, 0, stream>>>(W1, w1f);
    pack_w2<<<(HID * 16) / 256, 256, 0, stream>>>(W2, w2f);
    conv_to_afrag<<<dim3(KPAD / 64, BATCH / 4), dim3(64, 4), 0, stream>>>(x, cw, afrag);
    mlp_fused<<<BATCH / 128, 256, 0, stream>>>(afrag, w1f, b1, w2f, b2, out);
}
